// PeakDetectionGRU_17892833755531
// MI455X (gfx1250) — matrix operand order repaired and checked by reading, not compiled, not run on hardware
//
#include <hip/hip_runtime.h>
#include <hip/hip_bf16.h>

#define T_LEN 65536
#define HDIM 128
#define G3 384
#define MIN_DIST 40

typedef __attribute__((ext_vector_type(16))) _Float16 v16h;
typedef __attribute__((ext_vector_type(8)))  _Float16 v8h;
typedef __attribute__((ext_vector_type(8)))  float    v8f;

__device__ __forceinline__ float sig_f(float x) {
    x = fminf(30.f, fmaxf(-30.f, x));
    return 1.f / (1.f + __expf(-x));
}
__device__ __forceinline__ float tanh_f(float x) {
    x = fminf(15.f, fmaxf(-15.f, x));
    float e = __expf(-2.f * x);
    return (1.f - e) / (1.f + e);
}

// One block per direction (blockIdx.x: 0=fwd, 1=bwd), 256 threads = 8 waves.
// Wave w owns output rows [48w, 48w+48): 3 WMMA M-tiles x 4 K-chunks.
// Whh (f16) lives in VGPRs as pre-swizzled A operands; h broadcast via LDS as B.
__global__ __launch_bounds__(256) void gru_kernel(
    const float* __restrict__ sig,
    const float* __restrict__ Wih_f, const float* __restrict__ Whh_f,
    const float* __restrict__ bih_f, const float* __restrict__ bhh_f,
    const float* __restrict__ Wih_b, const float* __restrict__ Whh_b,
    const float* __restrict__ bih_b, const float* __restrict__ bhh_b,
    const float* __restrict__ w_out,
    float* __restrict__ logitF, float* __restrict__ logitB)
{
    const int dir  = blockIdx.x;
    const int tid  = threadIdx.x;
    const int lane = tid & 31;
    const int wave = tid >> 5;

    const float* Wih  = dir ? Wih_b : Wih_f;
    const float* Whh  = dir ? Whh_b : Whh_f;
    const float* bih  = dir ? bih_b : bih_f;
    const float* bhh  = dir ? bhh_b : bhh_f;
    const float* wpar = w_out + dir * HDIM;
    float* logit      = dir ? logitB : logitF;

    __shared__ alignas(32) _Float16 sh_h[HDIM];   // h as f16 (B operand source)
    __shared__ alignas(16) float    sh_y[G3];     // Whh*h result
    __shared__ alignas(16) float    sh_p[HDIM];   // per-element logit products

    // ---- Load A operands (Whh rows) into VGPRs in the documented 16-bit
    // A 16x32 layout: lanes 0-15 (M=row): e0..7 -> K k0..k0+7, e8..15 -> K k0+16..k0+23
    //                 lanes 16-31: same M, K shifted by +8.
    v16h A[12];
    {
        const int klo = (lane < 16) ? 0 : 8;
        const int mr  = lane & 15;
#pragma unroll
        for (int i = 0; i < 3; ++i) {
            const int row = wave * 48 + i * 16 + mr;
#pragma unroll
            for (int c = 0; c < 4; ++c) {
                const int k0 = c * 32;
                v16h a;
#pragma unroll
                for (int e = 0; e < 8; ++e) {
                    a[e]     = (_Float16)Whh[row * HDIM + k0 + klo + e];
                    a[8 + e] = (_Float16)Whh[row * HDIM + k0 + 16 + klo + e];
                }
                A[i * 4 + c] = a;
            }
        }
    }
    // Register barrier: make the packed-f16 A operands opaque so the compiler
    // keeps them resident instead of rematerializing v_cvt_pk_f16_f32 (and the
    // f32 weight copies) inside the recurrence loop.
#pragma unroll
    for (int i = 0; i < 12; ++i)
        asm volatile("" : "+v"(A[i]));

    // ---- Gate-thread private constants (j = tid < 128)
    float wr = 0, wz = 0, wn = 0, br = 0, bz = 0, bn = 0;
    float cr = 0, cz = 0, cn = 0, wo = 0, h = 0.f;
    if (tid < HDIM) {
        wr = Wih[tid]; wz = Wih[HDIM + tid]; wn = Wih[2 * HDIM + tid];
        br = bih[tid]; bz = bih[HDIM + tid]; bn = bih[2 * HDIM + tid];
        cr = bhh[tid]; cz = bhh[HDIM + tid]; cn = bhh[2 * HDIM + tid];
        wo = wpar[tid];
        sh_h[tid] = (_Float16)0.f;
    }
    __syncthreads();

    const int bhalf = (lane >> 4) << 3;  // B layout: lanes 0-15 K=k0..k0+7,k0+16..k0+23; lanes 16-31 K shifted by +8

    for (int t = 0; t < T_LEN; ++t) {
        const int tt = dir ? (T_LEN - 1 - t) : t;

        float s_t = 0.f;
        if (tid < HDIM) s_t = sig[tt];  // issued early; consumed after barrier

        // Wave 4: reduce previous step's logit (off critical path)
        if (wave == 4 && t > 0) {
            float4 v = *(const float4*)&sh_p[lane * 4];
            float sum = v.x + v.y + v.z + v.w;
#pragma unroll
            for (int off = 16; off; off >>= 1) sum += __shfl_xor(sum, off, 32);
            if (lane == 0) logit[dir ? (T_LEN - t) : (t - 1)] = sum;
        }

        // ---- Build broadcast B operands from h (f16 in LDS)
        v16h Bv[4];
#pragma unroll
        for (int c = 0; c < 4; ++c)
            Bv[c] = __builtin_shufflevector(*(const v8h*)(sh_h + c * 32 + bhalf), *(const v8h*)(sh_h + c * 32 + 16 + bhalf), 0, 1, 2, 3, 4, 5, 6, 7, 8, 9, 10, 11, 12, 13, 14, 15);

        // ---- 3 tiles x 4 K-chunks of v_wmma_f32_16x16x32_f16
#pragma unroll
        for (int i = 0; i < 3; ++i) {
            v8f acc = {0.f, 0.f, 0.f, 0.f, 0.f, 0.f, 0.f, 0.f};
#pragma unroll
            for (int c = 0; c < 4; ++c)
                acc = __builtin_amdgcn_wmma_f32_16x16x32_f16(
                    false, A[i * 4 + c], false, Bv[c], (short)0, acc, false, false);
            // D column N=0: lane 0 holds M=r in VGPR r, lane 16 holds M=8+r
            if ((lane & 15) == 0) {
                const int m0 = wave * 48 + i * 16 + ((lane >> 4) << 3);
                *(float4*)&sh_y[m0]     = make_float4(acc[0], acc[1], acc[2], acc[3]);
                *(float4*)&sh_y[m0 + 4] = make_float4(acc[4], acc[5], acc[6], acc[7]);
            }
        }
        __syncthreads();

        // ---- Gates (exact f32), h update, logit partial products
        if (tid < HDIM) {
            const float yr = sh_y[tid] + cr;
            const float yz = sh_y[HDIM + tid] + cz;
            const float yn = sh_y[2 * HDIM + tid] + cn;
            const float r = sig_f(fmaf(s_t, wr, br) + yr);
            const float z = sig_f(fmaf(s_t, wz, bz) + yz);
            const float n = tanh_f(fmaf(s_t, wn, bn) + r * yn);
            h = (1.f - z) * n + z * h;
            sh_h[tid] = (_Float16)h;
            sh_p[tid] = wo * h;
        }
        __syncthreads();
    }

    // Final step's logit
    if (wave == 4) {
        float4 v = *(const float4*)&sh_p[lane * 4];
        float sum = v.x + v.y + v.z + v.w;
#pragma unroll
        for (int off = 16; off; off >>= 1) sum += __shfl_xor(sum, off, 32);
        if (lane == 0) logit[dir ? 0 : (T_LEN - 1)] = sum;
    }
}

__global__ __launch_bounds__(256) void probs_kernel(
    const float* __restrict__ logitF, const float* __restrict__ logitB,
    const float* __restrict__ b_out, float* __restrict__ out,
    unsigned* __restrict__ maskw)
{
    const int i = blockIdx.x * 256 + threadIdx.x;
    const float l = logitF[i] + logitB[i] + b_out[0];
    const float p = 1.f / (1.f + __expf(-l));
    out[i] = p;                 // probs
    out[T_LEN + i] = 0.f;       // kept init
    const unsigned long long b = __ballot(p > 0.5f);
    if ((threadIdx.x & 31) == 0) maskw[i >> 5] = (unsigned)b;
}

__global__ __launch_bounds__(256) void nms_kernel(
    const unsigned* __restrict__ maskw, const float* __restrict__ probs,
    float* __restrict__ kept)
{
    __shared__ unsigned mw[T_LEN / 32];
    for (int i = threadIdx.x; i < T_LEN / 32; i += blockDim.x) mw[i] = maskw[i];
    __syncthreads();
    if (threadIdx.x == 0) {
        int win_end = -1, best_idx = 0;
        float best_p = -1e30f;
        bool have = false;
        for (int w = 0; w < T_LEN / 32; ++w) {
            unsigned word = mw[w];
            while (word) {                       // only visit peak positions
                const int b = __ffs(word) - 1;
                word &= word - 1;
                const int t = (w << 5) + b;
                const float p = probs[t];
                if (t >= win_end) {              // new window: emit previous best
                    if (have) kept[best_idx] = 1.0f;
                    win_end = t + MIN_DIST; best_idx = t; best_p = p; have = true;
                } else if (p > best_p) {
                    best_idx = t; best_p = p;
                }
            }
        }
        if (have) kept[best_idx] = 1.0f;
    }
}

extern "C" void kernel_launch(void* const* d_in, const int* in_sizes, int n_in,
                              void* d_out, int out_size, void* d_ws, size_t ws_size,
                              hipStream_t stream) {
    const float* sigp  = (const float*)d_in[0];
    const float* Wih_f = (const float*)d_in[1];
    const float* Whh_f = (const float*)d_in[2];
    const float* bih_f = (const float*)d_in[3];
    const float* bhh_f = (const float*)d_in[4];
    const float* Wih_b = (const float*)d_in[5];
    const float* Whh_b = (const float*)d_in[6];
    const float* bih_b = (const float*)d_in[7];
    const float* bhh_b = (const float*)d_in[8];
    const float* w_out = (const float*)d_in[9];
    const float* b_out = (const float*)d_in[10];

    float* out    = (float*)d_out;
    float* logitF = (float*)d_ws;
    float* logitB = logitF + T_LEN;
    unsigned* mw  = (unsigned*)(logitB + T_LEN);

    gru_kernel<<<2, 256, 0, stream>>>(sigp, Wih_f, Whh_f, bih_f, bhh_f,
                                      Wih_b, Whh_b, bih_b, bhh_b,
                                      w_out, logitF, logitB);
    probs_kernel<<<T_LEN / 256, 256, 0, stream>>>(logitF, logitB, b_out, out, mw);
    nms_kernel<<<1, 256, 0, stream>>>(mw, out, out + T_LEN);
}
